// SGRUCell_56899726737664
// MI455X (gfx1250) — compile-verified, matrix operand order checked by reading, not run on hardware
//
#include <hip/hip_runtime.h>
#include <stdint.h>

// ---------------- CDNA5 WMMA types ----------------
typedef __attribute__((ext_vector_type(16))) __bf16 v16bf;
typedef __attribute__((ext_vector_type(8)))  float  v8f;

#define BATCH   4096
#define UNITS   1024
#define KDIM    1024          // inner GEMM dim == UNITS
#define RSTRIDE 4096          // 4*UNITS columns in recurrent_kernel / kernel_c

// ---------------- CDNA5 async global->LDS path (guarded) ----------------
#if __has_builtin(__builtin_amdgcn_global_load_async_to_lds_b128)
#define HAVE_ASYNC_LDS 1
#else
#define HAVE_ASYNC_LDS 0
#endif

#if HAVE_ASYNC_LDS
#if __has_builtin(__builtin_amdgcn_s_wait_asynccnt)
#define WAIT_ASYNC(n) __builtin_amdgcn_s_wait_asynccnt(n)
#else
#define WAIT_ASYNC(n) asm volatile("s_wait_asynccnt " #n ::: "memory")
#endif
#else
#define WAIT_ASYNC(n) ((void)0)
#endif

// exact parameter type from hipcc diagnostic: 'int __vector(4) __device__ *'
typedef int v4i __attribute__((vector_size(16)));
#define AS_GLOBAL __attribute__((address_space(1)))
#define AS_LOCAL  __attribute__((address_space(3)))

__device__ __forceinline__ void cp16_to_lds(const uint16_t* g, uint4* l) {
#if HAVE_ASYNC_LDS
    __builtin_amdgcn_global_load_async_to_lds_b128(
        (AS_GLOBAL v4i*)(v4i*)g,    // global source (strip const, retype, addrspacecast)
        (AS_LOCAL  v4i*)(v4i*)l,    // LDS destination
        /*offset=*/0, /*cpol=*/0);
#else
    *l = *(const uint4*)g;
#endif
}

// round-to-nearest-even f32 -> bf16 bits
__device__ __forceinline__ uint16_t f32_to_bf16(float f) {
    uint32_t u = __float_as_uint(f);
    uint32_t r = u + 0x7FFFu + ((u >> 16) & 1u);
    return (uint16_t)(r >> 16);
}

__device__ __forceinline__ float sigmoidf_(float x) {
    return 1.0f / (1.0f + __expf(-x));
}

// ---------------- conversion kernels ----------------
__global__ void cvt_h_kernel(const float* __restrict__ src,
                             uint16_t* __restrict__ dst, int n) {
    int i = blockIdx.x * blockDim.x + threadIdx.x;
    if (i < n) dst[i] = f32_to_bf16(src[i]);
}

// R: [K=1024][4096] f32 -> Rt: [4096][1024] bf16, LDS-tiled transpose (coalesced both ways)
__global__ __launch_bounds__(256) void cvt_Rt_kernel(const float* __restrict__ R,
                                                     uint16_t* __restrict__ Rt) {
    __shared__ uint16_t tile[32][34];   // +2 pad to avoid bank conflicts
    const int tx = threadIdx.x & 31;
    const int ty = threadIdx.x >> 5;    // 0..7
    const int n0 = blockIdx.x * 32;     // column of R (gate index)
    const int k0 = blockIdx.y * 32;     // row of R (k index)
    for (int i = 0; i < 4; ++i) {
        int k = ty + i * 8;
        tile[k][tx] = f32_to_bf16(R[(size_t)(k0 + k) * RSTRIDE + n0 + tx]);
    }
    __syncthreads();
    for (int i = 0; i < 4; ++i) {
        int n = ty + i * 8;
        Rt[(size_t)(n0 + n) * KDIM + k0 + tx] = tile[tx][n];
    }
}

// ---------------- fused WMMA GEMM + GRU epilogue ----------------
// Block tile 128(M) x 64(N), K-step 32, 8 waves (4x2), wave tile 32x32 (2x2 WMMA).
// Double-buffered LDS, async global->LDS copies tracked with ASYNCcnt.
// MODE 1: N=2048 -> z (cols<1024, store z f32) and r (cols>=1024, store g=r*h bf16)
// MODE 2: N=1024 -> hh=tanh, h_new = z*h+(1-z)*hh -> d_out(h) f32 + bf16 ws
// MODE 3: N=1024 -> o = tanh -> d_out(o) f32
template <int MODE>
__global__ __launch_bounds__(256, 2) void gemm_gru_kernel(
    const uint16_t* __restrict__ A,     // [BATCH][1024] bf16
    const uint16_t* __restrict__ Rt,    // [4096][1024] bf16
    int n_off,                          // gate-column offset for this stage
    const int*   __restrict__ idx,      // inputs [BATCH]
    const float* __restrict__ kc,       // kernel_c [8000][4096]
    const float* __restrict__ h_in,     // h_tm1 f32
    const float* __restrict__ bias_z,   // [1024]
    const float* __restrict__ bias,     // [3072]
    float*       __restrict__ zbuf,     // [BATCH][1024] f32
    uint16_t*    __restrict__ out_bf,   // g or h_new bf16
    float*       __restrict__ out_f32)  // h_new or o (f32)
{
    __shared__ uint4 sA[2][512];    // 2 x (128 rows x 32 bf16) = 16 KB
    __shared__ uint4 sB[2][256];    // 2 x (64 rows  x 32 bf16) =  8 KB

    const int tid  = threadIdx.x;
    const int lane = tid & 31;
    const int wid  = tid >> 5;
    const int wm   = wid & 3;        // 0..3 -> M sub-tile
    const int wn   = wid >> 2;       // 0..1 -> N sub-tile
    const int lr   = lane & 15;      // row/col within fragment
    const int lh   = lane >> 4;      // lane half selects K sub-pattern

    const int m0 = blockIdx.x * 128;
    const int n0 = blockIdx.y * 64;

    // per-thread staging coordinates (each thread: 2 A pieces + 1 B piece, 16B each)
    const int ra = tid >> 2, cc = tid & 3;
    const uint16_t* gA0 = A  + (size_t)(m0 + ra)            * KDIM + cc * 8;
    const uint16_t* gA1 = A  + (size_t)(m0 + 64 + ra)       * KDIM + cc * 8;
    const uint16_t* gB  = Rt + (size_t)(n_off + n0 + ra)    * KDIM + cc * 8;

    v8f acc[2][2];
    for (int i = 0; i < 2; ++i)
        for (int j = 0; j < 2; ++j)
            acc[i][j] = v8f{};

    auto issue = [&](int k0, int buf) {
        cp16_to_lds(gA0 + k0, &sA[buf][tid]);
        cp16_to_lds(gA1 + k0, &sA[buf][tid + 256]);
        cp16_to_lds(gB + k0, &sB[buf][tid]);
    };

    constexpr int NSTEP = KDIM / 32;
    issue(0, 0);
    for (int t = 0; t < NSTEP; ++t) {
        const int buf = t & 1;
        if (t + 1 < NSTEP) {
            issue((t + 1) * 32, buf ^ 1);   // prefetch next tile into other buffer
            WAIT_ASYNC(3);                  // this tile's 3 copies complete (in-order)
        } else {
            WAIT_ASYNC(0);
        }
        __syncthreads();                    // all waves' copies visible

        // fragments: per ISA 16-bit A 16x32 layout (VGPR0-3: K 8h..8h+7, VGPR4-7: K 16+8h..)
        union frag { uint4 q[2]; v16bf v; };
        frag af[2], bfr[2];
        for (int i = 0; i < 2; ++i) {
            const uint4* p = &sA[buf][(wm * 32 + i * 16 + lr) * 4];
            af[i].q[0] = p[lh];
            af[i].q[1] = p[lh + 2];
        }
        for (int j = 0; j < 2; ++j) {
            const uint4* p = &sB[buf][(wn * 32 + j * 16 + lr) * 4];
            bfr[j].q[0] = p[lh];
            bfr[j].q[1] = p[lh + 2];
        }
        for (int i = 0; i < 2; ++i)
            for (int j = 0; j < 2; ++j)
                acc[i][j] = __builtin_amdgcn_wmma_f32_16x16x32_bf16(
                    false, af[i].v, false, bfr[j].v,
                    (short)0, acc[i][j], false, false);
        __syncthreads();                    // release this buffer for overwrite
    }

    // --- fused epilogue: C/D layout -> lane holds n=lr, VGPR v holds m = v + 8*lh ---
    for (int i = 0; i < 2; ++i) {
        for (int j = 0; j < 2; ++j) {
            const int jn = n0 + wn * 32 + j * 16 + lr;     // stage-local column
            for (int v = 0; v < 8; ++v) {
                const int m = m0 + wm * 32 + i * 16 + lh * 8 + v;
                float x = acc[i][j][v];
                const int ch = idx[m];
                // kernel_c gather is stream-once: non-temporal, keep L2 for Rt/A
                const float c = __builtin_nontemporal_load(
                    kc + (size_t)ch * RSTRIDE + n_off + jn);
                if (MODE == 1) {
                    if (jn < UNITS) {                      // z gate
                        float z = sigmoidf_(x + c + bias_z[jn]);
                        zbuf[(size_t)m * UNITS + jn] = z;
                    } else {                               // r gate -> g = r*h
                        int u = jn - UNITS;
                        float r = sigmoidf_(x + c + bias[u]);
                        out_bf[(size_t)m * UNITS + u] =
                            f32_to_bf16(r * h_in[(size_t)m * UNITS + u]);
                    }
                } else if (MODE == 2) {                    // candidate + blend
                    float hh = tanhf(x + c + bias[UNITS + jn]);
                    float z  = zbuf[(size_t)m * UNITS + jn];
                    float hn = z * h_in[(size_t)m * UNITS + jn] + (1.0f - z) * hh;
                    __builtin_nontemporal_store(hn, out_f32 + (size_t)m * UNITS + jn);
                    out_bf[(size_t)m * UNITS + jn] = f32_to_bf16(hn);
                } else {                                   // output gate
                    float o = tanhf(x + c + bias[2 * UNITS + jn]);
                    __builtin_nontemporal_store(o, out_f32 + (size_t)m * UNITS + jn);
                }
            }
        }
    }
}

// ---------------- launcher ----------------
extern "C" void kernel_launch(void* const* d_in, const int* in_sizes, int n_in,
                              void* d_out, int out_size, void* d_ws, size_t ws_size,
                              hipStream_t stream) {
    (void)in_sizes; (void)n_in; (void)out_size; (void)ws_size;
    const int*   idx = (const int*)  d_in[0];
    const float* h   = (const float*)d_in[1];
    const float* R   = (const float*)d_in[2];
    const float* kc  = (const float*)d_in[3];
    const float* bz  = (const float*)d_in[4];
    const float* b   = (const float*)d_in[5];

    float* out_o = (float*)d_out;
    float* out_h = out_o + (size_t)BATCH * UNITS;

    char* ws = (char*)d_ws;
    uint16_t* Rt   = (uint16_t*)(ws);                         //  8 MB
    uint16_t* hb   = (uint16_t*)(ws + ((size_t) 8 << 20));    //  8 MB
    uint16_t* gb   = (uint16_t*)(ws + ((size_t)16 << 20));    //  8 MB
    uint16_t* hnb  = (uint16_t*)(ws + ((size_t)24 << 20));    //  8 MB
    float*    zbuf = (float*)   (ws + ((size_t)32 << 20));    // 16 MB

    cvt_h_kernel <<<(BATCH * UNITS) / 256, 256, 0, stream>>>(h, hb, BATCH * UNITS);
    cvt_Rt_kernel<<<dim3(RSTRIDE / 32, KDIM / 32), 256, 0, stream>>>(R, Rt);

    dim3 blk(256);
    // stage 1: [Rz|Rr] fused, N = 2048
    gemm_gru_kernel<1><<<dim3(BATCH / 128, 2048 / 64), blk, 0, stream>>>(
        hb, Rt, 0, idx, kc, h, bz, b, zbuf, gb, nullptr);
    // stage 2: Rh, N = 1024 (A = g = r*h)
    gemm_gru_kernel<2><<<dim3(BATCH / 128, 1024 / 64), blk, 0, stream>>>(
        gb, Rt, 2048, idx, kc, h, bz, b, zbuf, hnb, out_h);
    // stage 3: Ro, N = 1024 (A = h_new)
    gemm_gru_kernel<3><<<dim3(BATCH / 128, 1024 / 64), blk, 0, stream>>>(
        hnb, Rt, 3072, idx, kc, h, bz, b, zbuf, nullptr, out_o);
}
